// Net_17197049053679
// MI455X (gfx1250) — matrix operand order repaired and checked by reading, not compiled, not run on hardware
//
#include <hip/hip_runtime.h>
#include <hip/hip_bf16.h>
#include <cstdint>

// ---------------- types / constants ----------------
typedef _Float16 h16;
typedef __attribute__((ext_vector_type(8)))  _Float16 v8h;
typedef __attribute__((ext_vector_type(16))) _Float16 v16h;
typedef __attribute__((ext_vector_type(8)))  float    v8f;

constexpr int kB    = 128;     // graphs
constexpr int kNPG  = 32;      // nodes per graph (level 1)
constexpr int kSEQ  = 16;
constexpr int kCF   = 3;
constexpr int kCED  = 50;
constexpr int kCFS  = 50;
constexpr int kWED  = 300;
constexpr int kH    = 128;
constexpr int kWWIN = 12;      // MWL + CF - 1
constexpr int kLC   = 194;     // SEQ*WWIN + CF - 1
constexpr int kN    = 4096;    // B * NPG
constexpr int kE    = 16384;   // B * EPG
constexpr int kFIN  = 350;     // WED + CFS
constexpr int kFINP = 352;     // padded to 11*32 for WMMA K
constexpr int kG4   = 512;     // 4*H LSTM gates
constexpr float kBIG = 1000000.0f;

__device__ inline float sigm(float x) { return 1.0f / (1.0f + __expf(-x)); }

// ---------------- WMMA tile helpers (CDNA5 wave32, 16x16x32 f16) ----------------
// A (16x32 f16, row-major source, leading dim lda):
//   lanes 0-15 : row = row0+lane,    halves 0..7 = K kk+0..7,  8..15 = K kk+16..23
//   lanes 16-31: row = row0+lane-16, halves 0..7 = K kk+8..15, 8..15 = K kk+24..31
__device__ inline v16h ld_matA(const h16* __restrict__ A, int lda, int row0, int kk) {
  const int lane = threadIdx.x & 31;
  const h16* base = A + (size_t)(row0 + (lane & 15)) * lda + kk + ((lane & 16) ? 8 : 0);
  __builtin_prefetch(base + 32, 0, 0);       // next K-chunk -> global_prefetch_b8
  v8h a0 = *(const v8h*)(base);
  v8h a1 = *(const v8h*)(base + 16);
  v16h r;
#pragma unroll
  for (int i = 0; i < 8; ++i) { r[i] = a0[i]; r[8 + i] = a1[i]; }
  return r;
}

// B (32x16 f16) loaded from Bt = B^T stored row-major (N x K, leading dim ldb):
//   lanes 0-15 : col = col0+lane,    halves 0..7 = K kk+0..7,  8..15 = K kk+16..23
//   lanes 16-31: col = col0+lane-16, halves 0..7 = K kk+8..15, 8..15 = K kk+24..31
__device__ inline v16h ld_matB(const h16* __restrict__ Bt, int ldb, int col0, int kk) {
  const int lane = threadIdx.x & 31;
  const h16* base = Bt + (size_t)(col0 + (lane & 15)) * ldb + kk + ((lane & 16) ? 8 : 0);
  v8h b0 = *(const v8h*)(base);
  v8h b1 = *(const v8h*)(base + 16);
  v16h r;
#pragma unroll
  for (int i = 0; i < 8; ++i) { r[i] = b0[i]; r[8 + i] = b1[i]; }
  return r;
}

__device__ inline v8f wmma_f16(v16h a, v16h b, v8f c) {
  // 8 args: (neg_a, A, neg_b, B, c_mod, C, reuse_a, reuse_b)
  return __builtin_amdgcn_wmma_f32_16x16x32_f16(false, a, false, b, (short)0, c, false, false);
}

// C/D 16x16 f32: VGPR i -> lanes 0-15 row=i, lanes 16-31 row=8+i; col = lane&15.
__device__ inline void st_tileC(float* __restrict__ C, int ldc, int row0, int col0,
                                v8f acc, const float* __restrict__ bias) {
  const int lane = threadIdx.x & 31;
  const int c  = col0 + (lane & 15);
  const int r0 = row0 + ((lane & 16) ? 8 : 0);
  const float bv = bias ? bias[c] : 0.0f;
#pragma unroll
  for (int i = 0; i < 8; ++i) C[(size_t)(r0 + i) * ldc + c] = acc[i] + bv;
}

// ---------------- generic WMMA GEMM:  C(MxN,f32) = A(MxK,f16) @ Bt(NxK,f16)^T ----
// Register-blocked: each wave computes a 32x32 tile (4 accumulators, A/B reuse).
// grid: (N/32, M/128, 1), block 128 (4 waves stacked along M). EXEC all-ones.
__global__ __launch_bounds__(128) void k_wmma_gemm(
    const h16* __restrict__ A, const h16* __restrict__ Bt, float* __restrict__ C,
    int K, int lda, int ldb, int ldc)
{
  const int wave = threadIdx.x >> 5;
  const int row0 = blockIdx.y * 128 + wave * 32;
  const int col0 = blockIdx.x * 32;
  v8f acc00 = {}, acc01 = {}, acc10 = {}, acc11 = {};
  for (int kk = 0; kk < K; kk += 32) {
    v16h a0 = ld_matA(A, lda, row0,      kk);
    v16h a1 = ld_matA(A, lda, row0 + 16, kk);
    v16h b0 = ld_matB(Bt, ldb, col0,      kk);
    v16h b1 = ld_matB(Bt, ldb, col0 + 16, kk);
    acc00 = wmma_f16(a0, b0, acc00);
    acc01 = wmma_f16(a0, b1, acc01);
    acc10 = wmma_f16(a1, b0, acc10);
    acc11 = wmma_f16(a1, b1, acc11);
  }
  st_tileC(C, ldc, row0,      col0,      acc00, nullptr);
  st_tileC(C, ldc, row0,      col0 + 16, acc01, nullptr);
  st_tileC(C, ldc, row0 + 16, col0,      acc10, nullptr);
  st_tileC(C, ldc, row0 + 16, col0 + 16, acc11, nullptr);
}

// ---------------- fused LSTM step GEMM:  G = xi[t] @ Wih^T + h @ Whh^T + b -------
// 32x32 wave tiles; grid: (512/32, 4096/128, 2 dirs), block 128.
__global__ __launch_bounds__(128) void k_lstm_step(
    const h16* __restrict__ xi, const h16* __restrict__ hX,
    const h16* __restrict__ WihH, const h16* __restrict__ WhhH,
    const float* __restrict__ bF, const float* __restrict__ bB,
    float* __restrict__ G, int tf, int tb)
{
  const int dir = blockIdx.z;
  const h16* xt  = xi + (size_t)(dir ? tb : tf) * kN * kFINP;
  const h16* Wih = WihH + (size_t)dir * kG4 * kFINP;
  const h16* Whh = WhhH + (size_t)dir * kG4 * kH;
  const float* bias = dir ? bB : bF;
  float* Gd = G + (size_t)dir * kN * kG4;

  const int wave = threadIdx.x >> 5;
  const int row0 = blockIdx.y * 128 + wave * 32;
  const int col0 = blockIdx.x * 32;
  v8f acc00 = {}, acc01 = {}, acc10 = {}, acc11 = {};
#pragma unroll 4
  for (int kk = 0; kk < kFINP; kk += 32) {         // input projection, K=352
    v16h a0 = ld_matA(xt, kFINP, row0,      kk);
    v16h a1 = ld_matA(xt, kFINP, row0 + 16, kk);
    v16h b0 = ld_matB(Wih, kFINP, col0,      kk);
    v16h b1 = ld_matB(Wih, kFINP, col0 + 16, kk);
    acc00 = wmma_f16(a0, b0, acc00);
    acc01 = wmma_f16(a0, b1, acc01);
    acc10 = wmma_f16(a1, b0, acc10);
    acc11 = wmma_f16(a1, b1, acc11);
  }
  const h16* hA = hX + dir * kH;                    // h lives in columns dir*128..
#pragma unroll
  for (int kk = 0; kk < kH; kk += 32) {            // recurrent projection, K=128
    v16h a0 = ld_matA(hA, 2 * kH, row0,      kk);
    v16h a1 = ld_matA(hA, 2 * kH, row0 + 16, kk);
    v16h b0 = ld_matB(Whh, kH, col0,      kk);
    v16h b1 = ld_matB(Whh, kH, col0 + 16, kk);
    acc00 = wmma_f16(a0, b0, acc00);
    acc01 = wmma_f16(a0, b1, acc01);
    acc10 = wmma_f16(a1, b0, acc10);
    acc11 = wmma_f16(a1, b1, acc11);
  }
  st_tileC(Gd, kG4, row0,      col0,      acc00, bias);
  st_tileC(Gd, kG4, row0,      col0 + 16, acc01, bias);
  st_tileC(Gd, kG4, row0 + 16, col0,      acc10, bias);
  st_tileC(Gd, kG4, row0 + 16, col0 + 16, acc11, bias);
}

// ---------------- LSTM cell pointwise --------------------------------------------
__global__ void k_lstm_cell(const float* __restrict__ G, float* __restrict__ c,
                            h16* __restrict__ hX)
{
  int tid = blockIdx.x * blockDim.x + threadIdx.x;
  if (tid >= 2 * kN * kH) return;
  int dir = tid >> 19;            // kN*kH = 2^19
  int r   = (tid >> 7) & (kN - 1);
  int j   = tid & (kH - 1);
  const float* g = G + (size_t)dir * kN * kG4 + (size_t)r * kG4;
  float gi = g[j], gf = g[kH + j], gc = g[2 * kH + j], go = g[3 * kH + j];
  float cv = c[tid];
  cv = sigm(gf) * cv + sigm(gi) * tanhf(gc);
  float hv = sigm(go) * tanhf(cv);
  c[tid] = cv;
  hX[(size_t)r * (2 * kH) + dir * kH + j] = (h16)hv;
}

// ---------------- frontend: embeddings + char conv -------------------------------
__global__ void k_word_emb(const int* __restrict__ words, const float* __restrict__ emb,
                           h16* __restrict__ xi)
{
  int tid = blockIdx.x * blockDim.x + threadIdx.x;
  if (tid >= kSEQ * kN * kWED) return;
  int cidx = tid % kWED;
  int n = (tid / kWED) % kN;
  int t = tid / (kWED * kN);
  int w = words[n * kSEQ + t];
  xi[((size_t)t * kN + n) * kFINP + cidx] = (h16)emb[(size_t)w * kWED + cidx];
}

__global__ __launch_bounds__(256) void k_charconv(
    const int* __restrict__ chars, const float* __restrict__ emb,
    const float* __restrict__ cw, const float* __restrict__ cb,
    h16* __restrict__ xi)
{
  __shared__ float sw[kCFS * kCED * kCF];   // 7500
  __shared__ float se[100 * kCED];          // 5000
  for (int i = threadIdx.x; i < kCFS * kCED * kCF; i += 256) sw[i] = cw[i];
  for (int i = threadIdx.x; i < 100 * kCED; i += 256) se[i] = emb[i];
  __syncthreads();
  int id = blockIdx.x * 256 + threadIdx.x;       // N*SEQ*CFS = 3,276,800 exact
  if (id >= kN * kSEQ * kCFS) return;
  int f = id % kCFS;
  int t = (id / kCFS) % kSEQ;
  int n = id / (kCFS * kSEQ);
  const int* ch = chars + (size_t)n * kLC + t * kWWIN;
  float mx = -1e30f;
  for (int w = 0; w < kWWIN; ++w) {
    float s = 0.0f;
#pragma unroll
    for (int k = 0; k < kCF; ++k) {
      const float* er = se + ch[w + k] * kCED;
      const float* wr = sw + f * (kCED * kCF) + k;
#pragma unroll
      for (int cc = 0; cc < kCED; ++cc) s += er[cc] * wr[cc * kCF];
    }
    mx = fmaxf(mx, s);
  }
  float val = tanhf(mx + cb[f]);
  xi[((size_t)t * kN + n) * kFINP + kWED + f] = (h16)val;
}

// ---------------- small weight prep ----------------------------------------------
__global__ void k_cvt_pad(const float* __restrict__ src, h16* __restrict__ dst,
                          int rows, int cols, int ldd)
{
  int tid = blockIdx.x * blockDim.x + threadIdx.x;
  if (tid >= rows * ldd) return;
  int r = tid / ldd, c = tid % ldd;
  dst[tid] = (c < cols) ? (h16)src[(size_t)r * cols + c] : (h16)0.0f;
}

__global__ void k_cvt_transpose(const float* __restrict__ src, h16* __restrict__ dst,
                                int K, int Nc)   // src (K x Nc) -> dst (Nc x K) f16
{
  int tid = blockIdx.x * blockDim.x + threadIdx.x;
  if (tid >= K * Nc) return;
  int k = tid / Nc, n = tid % Nc;
  dst[(size_t)n * K + k] = (h16)src[tid];
}

// ---------------- graph machinery ------------------------------------------------
__global__ void k_edge_init(const int* __restrict__ eix, int* __restrict__ src,
                            int* __restrict__ dst, float* __restrict__ ew)
{
  int e = blockIdx.x * blockDim.x + threadIdx.x;
  if (e >= kE) return;
  src[e] = eix[e]; dst[e] = eix[kE + e]; ew[e] = 1.0f;
}

__global__ void k_eidx_init(const int* __restrict__ ent, const int* __restrict__ sent,
                            int* __restrict__ e1, int* __restrict__ e2, int* __restrict__ s)
{
  int g = threadIdx.x;
  if (g >= kB) return;
  e1[g] = ent[2 * g]; e2[g] = ent[2 * g + 1]; s[g] = sent[g];
}

__global__ void k_deg_init(float* __restrict__ deg, int n)
{ int i = blockIdx.x * blockDim.x + threadIdx.x; if (i < n) deg[i] = 1.0f; }   // self loop

__global__ void k_deg_acc(const int* __restrict__ dst, const float* __restrict__ ew,
                          float* __restrict__ deg)
{
  int e = blockIdx.x * blockDim.x + threadIdx.x;
  if (e >= kE) return;
  float w = ew[e];
  if (w != 0.0f) atomicAdd(&deg[dst[e]], w);
}

__global__ void k_dinv(const float* __restrict__ deg, float* __restrict__ dinv, int n)
{
  int i = blockIdx.x * blockDim.x + threadIdx.x;
  if (i >= n) return;
  float d = deg[i];
  dinv[i] = (d > 0.0f) ? rsqrtf(d) : 0.0f;
}

// out = xw * dinv^2 (self loop) + bias
__global__ void k_agg_init(const float* __restrict__ xw, const float* __restrict__ dinv,
                           const float* __restrict__ bias, float* __restrict__ out,
                           int n, int F)
{
  int tid = blockIdx.x * blockDim.x + threadIdx.x;
  if (tid >= n * F) return;
  int i = tid / F, f = tid - i * F;
  float dv = dinv[i];
  out[tid] = xw[tid] * dv * dv + bias[f];
}

__global__ void k_agg_edge(const float* __restrict__ xw, const float* __restrict__ dinv,
                           const int* __restrict__ src, const int* __restrict__ dst,
                           const float* __restrict__ ew, float* __restrict__ out, int F)
{
  int tid = blockIdx.x * blockDim.x + threadIdx.x;
  if (tid >= kE * F) return;
  int e = tid / F, f = tid - e * F;
  float w = ew[e];
  if (w == 0.0f) return;
  int sN = src[e], dN = dst[e];
  float coef = dinv[sN] * dinv[dN] * w;
  atomicAdd(&out[(size_t)dN * F + f], xw[(size_t)sN * F + f] * coef);
}

__global__ void k_relu(float* __restrict__ x, int cnt)
{
  int tid = blockIdx.x * blockDim.x + threadIdx.x;
  if (tid >= cnt) return;
  float v = x[tid];
  x[tid] = v > 0.0f ? v : 0.0f;
}

__global__ void k_dot_col(const float* __restrict__ x, const float* __restrict__ w,
                          float* __restrict__ out, int n)
{
  int i = blockIdx.x * blockDim.x + threadIdx.x;
  if (i >= n) return;
  const float* row = x + (size_t)i * kH;
  float s = 0.0f;
#pragma unroll 8
  for (int j = 0; j < kH; ++j) s += row[j] * w[j];
  out[i] = s;
}

__global__ void k_override(float* __restrict__ score, const int* __restrict__ e1,
                           const int* __restrict__ e2, const int* __restrict__ s)
{
  int g = threadIdx.x;
  if (g >= kB) return;
  score[e1[g]] = kBIG; score[e2[g]] = kBIG; score[s[g]] = kBIG;
}

// per-graph selection top-k (descending, ties -> lowest index), one thread / graph
__global__ void k_topk(const float* __restrict__ score, int* __restrict__ perm,
                       int npg, int k)
{
  int g = blockIdx.x * blockDim.x + threadIdx.x;
  if (g >= kB) return;
  int base = g * npg;
  unsigned used = 0;
  for (int i = 0; i < k; ++i) {
    float best = -3.4e38f; int bi = 0;
    for (int j = 0; j < npg; ++j) {
      if (used & (1u << j)) continue;
      float sv = score[base + j];
      if (sv > best) { best = sv; bi = j; }
    }
    used |= 1u << bi;
    perm[g * k + i] = base + bi;
  }
}

__global__ void k_zero_int2(int* __restrict__ a, int* __restrict__ b, int n)
{
  int i = blockIdx.x * blockDim.x + threadIdx.x;
  if (i < n) { a[i] = 0; b[i] = 0; }
}

__global__ void k_pool_scatter(const int* __restrict__ perm, int* __restrict__ mapping,
                               int* __restrict__ kept, int cnt)
{
  int i = blockIdx.x * blockDim.x + threadIdx.x;
  if (i >= cnt) return;
  int p = perm[i];
  mapping[p] = i; kept[p] = 1;
}

__global__ void k_pool_gather(const float* __restrict__ x, const float* __restrict__ score,
                              const int* __restrict__ perm, float* __restrict__ px,
                              h16* __restrict__ pxh, int cnt)
{
  int tid = blockIdx.x * blockDim.x + threadIdx.x;
  if (tid >= cnt) return;
  int i = tid / kH, f = tid - i * kH;
  int p = perm[i];
  float v = x[(size_t)p * kH + f] * tanhf(score[p]);
  px[tid] = v;
  pxh[tid] = (h16)v;
}

__global__ void k_edge_remap(int* __restrict__ src, int* __restrict__ dst,
                             float* __restrict__ ew, const int* __restrict__ mapping,
                             const int* __restrict__ kept)
{
  int e = blockIdx.x * blockDim.x + threadIdx.x;
  if (e >= kE) return;
  int sN = src[e], dN = dst[e];
  float w = ew[e];
  if (!(kept[sN] && kept[dN])) w = 0.0f;
  src[e] = mapping[sN]; dst[e] = mapping[dN]; ew[e] = w;
}

__global__ void k_e_update(int* __restrict__ e1, int* __restrict__ e2,
                           int* __restrict__ s, int k)
{
  int g = threadIdx.x;
  if (g >= kB) return;
  e1[g] = g * k; e2[g] = g * k + 1; s[g] = g * k + 2;
}

__global__ void k_readout(const float* __restrict__ px, float* __restrict__ xr, int k)
{
  int tid = blockIdx.x * blockDim.x + threadIdx.x;
  if (tid >= kB * kH) return;
  int g = tid / kH, f = tid - g * kH;
  float mx = -1e30f, sm = 0.0f;
  for (int i = 0; i < k; ++i) {
    float v = px[((size_t)g * k + i) * kH + f];
    mx = fmaxf(mx, v); sm += v;
  }
  xr[g * 256 + f] = mx;
  xr[g * 256 + kH + f] = sm / (float)k;
}

// out (128 x 1408): [e1x1|e1x2|e1x3 | e2.. | s.. | x1r+x2r+x3r]
__global__ void k_assembly(const float* __restrict__ px1, const float* __restrict__ px2,
                           const float* __restrict__ px3, const float* __restrict__ xr1,
                           const float* __restrict__ xr2, const float* __restrict__ xr3,
                           float* __restrict__ out)
{
  int tid = blockIdx.x * blockDim.x + threadIdx.x;
  if (tid >= kB * 1408) return;
  int g = tid / 1408, c = tid - g * 1408;
  float val;
  if (c < 1152) {
    int part = c / 384;        // 0:e1 1:e2 2:sent
    int rem  = c - part * 384;
    int lvl  = rem / kH;       // 0,1,2
    int f    = rem - lvl * kH;
    const float* px = (lvl == 0) ? px1 : (lvl == 1) ? px2 : px3;
    int kk = (lvl == 0) ? 16 : (lvl == 1) ? 8 : 4;
    val = px[((size_t)g * kk + part) * kH + f];
  } else {
    int f = c - 1152;          // 0..255
    val = xr1[g * 256 + f] + xr2[g * 256 + f] + xr3[g * 256 + f];
  }
  out[tid] = val;
}

// ---------------- host orchestration ----------------------------------------------
extern "C" void kernel_launch(void* const* d_in, const int* in_sizes, int n_in,
                              void* d_out, int out_size, void* d_ws, size_t ws_size,
                              hipStream_t stream)
{
  (void)in_sizes; (void)n_in; (void)out_size; (void)ws_size;
  const int*   words    = (const int*)d_in[0];
  const int*   chars    = (const int*)d_in[1];
  const int*   edge_ix  = (const int*)d_in[2];
  const int*   ent_ix   = (const int*)d_in[4];
  const int*   sent_ix  = (const int*)d_in[5];
  const float* word_emb = (const float*)d_in[6];
  const float* char_emb = (const float*)d_in[7];
  const float* conv_w   = (const float*)d_in[8];
  const float* conv_b   = (const float*)d_in[9];
  const float* Wih_f = (const float*)d_in[10]; const float* Whh_f = (const float*)d_in[11];
  const float* b_f   = (const float*)d_in[12];
  const float* Wih_b = (const float*)d_in[13]; const float* Whh_b = (const float*)d_in[14];
  const float* b_b   = (const float*)d_in[15];
  const float* W1 = (const float*)d_in[16]; const float* b1  = (const float*)d_in[17];
  const float* Wp1= (const float*)d_in[18]; const float* bp1 = (const float*)d_in[19];
  const float* W2 = (const float*)d_in[20]; const float* b2  = (const float*)d_in[21];
  const float* Wp2= (const float*)d_in[22]; const float* bp2 = (const float*)d_in[23];
  const float* W3 = (const float*)d_in[24]; const float* b3  = (const float*)d_in[25];
  const float* Wp3= (const float*)d_in[26]; const float* bp3 = (const float*)d_in[27];
  float* out = (float*)d_out;

  char* wsp = (char*)d_ws;
  auto alloc = [&](size_t bytes) -> void* {
    void* p = (void*)wsp; wsp += (bytes + 255) & ~size_t(255); return p;
  };
  h16*   xi    = (h16*)alloc((size_t)kSEQ * kN * kFINP * 2);   // 46 MB
  h16*   WihH  = (h16*)alloc((size_t)2 * kG4 * kFINP * 2);
  h16*   WhhH  = (h16*)alloc((size_t)2 * kG4 * kH * 2);
  h16*   xlstm = (h16*)alloc((size_t)kN * 2 * kH * 2);         // h state / LSTM output
  float* cst   = (float*)alloc((size_t)2 * kN * kH * 4);
  float* G     = (float*)alloc((size_t)2 * kN * kG4 * 4);
  h16*   W1t   = (h16*)alloc((size_t)kH * 2 * kH * 2);
  h16*   W2t   = (h16*)alloc((size_t)kH * kH * 2);
  h16*   W3t   = (h16*)alloc((size_t)kH * kH * 2);
  int*   srcA  = (int*)alloc(kE * 4);
  int*   dstA  = (int*)alloc(kE * 4);
  float* ew    = (float*)alloc(kE * 4);
  int*   e1    = (int*)alloc(kB * 4);
  int*   e2    = (int*)alloc(kB * 4);
  int*   sA    = (int*)alloc(kB * 4);
  float* deg   = (float*)alloc(kN * 4);
  float* dinv  = (float*)alloc(kN * 4);
  float* xw    = (float*)alloc((size_t)kN * kH * 4);
  float* xbuf  = (float*)alloc((size_t)kN * kH * 4);
  float* sdot  = (float*)alloc(kN * 4);
  float* score = (float*)alloc(kN * 4);
  int*   perm  = (int*)alloc(2048 * 4);
  int*   mapg  = (int*)alloc(kN * 4);
  int*   kept  = (int*)alloc(kN * 4);
  float* px1   = (float*)alloc((size_t)2048 * kH * 4);
  h16*   px1h  = (h16*)alloc((size_t)2048 * kH * 2);
  float* px2   = (float*)alloc((size_t)1024 * kH * 4);
  h16*   px2h  = (h16*)alloc((size_t)1024 * kH * 2);
  float* px3   = (float*)alloc((size_t)512 * kH * 4);
  h16*   px3h  = (h16*)alloc((size_t)512 * kH * 2);
  float* xr1   = (float*)alloc(kB * 256 * 4);
  float* xr2   = (float*)alloc(kB * 256 * 4);
  float* xr3   = (float*)alloc(kB * 256 * 4);

  auto g1 = [](long long n) { return dim3((unsigned)((n + 255) / 256)); };

  // ---- weight prep (f16 / transposed-for-WMMA copies) ----
  k_cvt_pad<<<g1(kG4 * kFINP), 256, 0, stream>>>(Wih_f, WihH, kG4, kFIN, kFINP);
  k_cvt_pad<<<g1(kG4 * kFINP), 256, 0, stream>>>(Wih_b, WihH + (size_t)kG4 * kFINP, kG4, kFIN, kFINP);
  k_cvt_pad<<<g1(kG4 * kH), 256, 0, stream>>>(Whh_f, WhhH, kG4, kH, kH);
  k_cvt_pad<<<g1(kG4 * kH), 256, 0, stream>>>(Whh_b, WhhH + (size_t)kG4 * kH, kG4, kH, kH);
  k_cvt_transpose<<<g1(2 * kH * kH), 256, 0, stream>>>(W1, W1t, 2 * kH, kH);
  k_cvt_transpose<<<g1(kH * kH), 256, 0, stream>>>(W2, W2t, kH, kH);
  k_cvt_transpose<<<g1(kH * kH), 256, 0, stream>>>(W3, W3t, kH, kH);

  // ---- frontend: xi = [word_emb | tanh(maxpool(charconv))] in f16, K padded ----
  hipMemsetAsync(xi, 0, (size_t)kSEQ * kN * kFINP * 2, stream);
  k_word_emb<<<g1((long long)kSEQ * kN * kWED), 256, 0, stream>>>(words, word_emb, xi);
  k_charconv<<<g1((long long)kN * kSEQ * kCFS), 256, 0, stream>>>(chars, char_emb, conv_w, conv_b, xi);

  // ---- BiLSTM: 16 fused WMMA-GEMM steps, both directions via grid.z ----
  hipMemsetAsync(xlstm, 0, (size_t)kN * 2 * kH * 2, stream);
  hipMemsetAsync(cst, 0, (size_t)2 * kN * kH * 4, stream);
  for (int t = 0; t < kSEQ; ++t) {
    dim3 gs(kG4 / 32, kN / 128, 2);
    k_lstm_step<<<gs, 128, 0, stream>>>(xi, xlstm, WihH, WhhH, b_f, b_b, G, t, kSEQ - 1 - t);
    k_lstm_cell<<<g1(2 * kN * kH), 256, 0, stream>>>(G, cst, xlstm);
  }

  // ---- graph init ----
  k_edge_init<<<g1(kE), 256, 0, stream>>>(edge_ix, srcA, dstA, ew);
  k_eidx_init<<<1, kB, 0, stream>>>(ent_ix, sent_ix, e1, e2, sA);

  // ---- one GCN + top-k-pool level ----
  auto run_level = [&](int n, int npg, int k, const h16* Ain, int K, const h16* Wt,
                       const float* bL, const float* Wp, const float* bp,
                       float* px, h16* pxh, float* xr) {
    // symmetric normalization
    k_deg_init<<<g1(n), 256, 0, stream>>>(deg, n);
    k_deg_acc<<<g1(kE), 256, 0, stream>>>(dstA, ew, deg);
    k_dinv<<<g1(n), 256, 0, stream>>>(deg, dinv, n);
    // xw = x @ W  (WMMA, 32x32 wave tiles)
    dim3 gg(kH / 32, n / 128, 1);
    k_wmma_gemm<<<gg, 128, 0, stream>>>(Ain, Wt, xw, K, K, K, kH);
    // aggregate + relu
    k_agg_init<<<g1((long long)n * kH), 256, 0, stream>>>(xw, dinv, bL, xbuf, n, kH);
    k_agg_edge<<<g1((long long)kE * kH), 256, 0, stream>>>(xw, dinv, srcA, dstA, ew, xbuf, kH);
    k_relu<<<g1((long long)n * kH), 256, 0, stream>>>(xbuf, n * kH);
    // pooling score GCN (F=1)
    k_dot_col<<<g1(n), 256, 0, stream>>>(xbuf, Wp, sdot, n);
    k_agg_init<<<g1(n), 256, 0, stream>>>(sdot, dinv, bp, score, n, 1);
    k_agg_edge<<<g1(kE), 256, 0, stream>>>(sdot, dinv, srcA, dstA, ew, score, 1);
    k_override<<<1, kB, 0, stream>>>(score, e1, e2, sA);
    // top-k + pooled node set
    k_topk<<<1, kB, 0, stream>>>(score, perm, npg, k);
    int nk = kB * k;
    k_zero_int2<<<g1(n), 256, 0, stream>>>(mapg, kept, n);
    k_pool_scatter<<<g1(nk), 256, 0, stream>>>(perm, mapg, kept, nk);
    k_pool_gather<<<g1((long long)nk * kH), 256, 0, stream>>>(xbuf, score, perm, px, pxh, nk * kH);
    k_edge_remap<<<g1(kE), 256, 0, stream>>>(srcA, dstA, ew, mapg, kept);
    k_e_update<<<1, kB, 0, stream>>>(e1, e2, sA, k);
    k_readout<<<g1(kB * kH), 256, 0, stream>>>(px, xr, k);
  };

  run_level(kN,   kNPG, 16, xlstm, 2 * kH, W1t, b1, Wp1, bp1, px1, px1h, xr1);
  run_level(2048, 16,   8,  px1h,  kH,     W2t, b2, Wp2, bp2, px2, px2h, xr2);
  run_level(1024, 8,    4,  px2h,  kH,     W3t, b3, Wp3, bp3, px3, px3h, xr3);

  k_assembly<<<g1(kB * 1408), 256, 0, stream>>>(px1, px2, px3, xr1, xr2, xr3, out);
}
